// MyEncoderLayer_24103356465339
// MI455X (gfx1250) — matrix operand order repaired and checked by reading, not compiled, not run on hardware
//
#include <hip/hip_runtime.h>
#include <hip/hip_bf16.h>

// ---------------- problem constants ----------------
#define EMB   256
#define ENC   256
#define BATCH 4
#define SEQ   4096
#define NROWS (BATCH * SEQ)     // 16384 rows of [256]
#define WELEM (ENC * EMB)       // 65536 per weight matrix

typedef __attribute__((ext_vector_type(16))) _Float16 v16h;
typedef __attribute__((ext_vector_type(8)))  float    v8f;
typedef __attribute__((ext_vector_type(4)))  unsigned v4u;
typedef __attribute__((ext_vector_type(8)))  unsigned v8u;

// 8-VGPR f16 fragment (16 halves / lane), loadable as two 16B chunks
union F16Frag {
    v16h v;
    _Float16 h[16];
    uint4 q[2];
};

static __device__ __forceinline__ v8f wmma_f16(const F16Frag& a, const F16Frag& b, v8f c) {
    return __builtin_amdgcn_wmma_f32_16x16x32_f16(false, a.v, false, b.v,
                                                  (short)0, c, false, false);
}

// generic -> LDS(AS3) byte offset
typedef __attribute__((address_space(3))) char lds_char;
static __device__ __forceinline__ unsigned lds_off_of(const void* p) {
    return (unsigned)(uintptr_t)(lds_char*)p;
}

// Issue a 2-D TDM load: tile_d0 x tile_d1 elements (f16, data_size code 1)
// from row-major tensor with dim-0 stride `stride0` (elements) into LDS
// (rows stored back-to-back: tile row stride = tile_d0 elements).
// D# layout per CDNA5 ISA 8.3/8.4; groups 2/3 omitted (2-D tensor -> NULL).
static __device__ __forceinline__ void tdm_load_2d_f16(
    unsigned lds_byte_off, const void* gptr,
    unsigned tensor_d0, unsigned tensor_d1,
    unsigned tile_d0, unsigned tile_d1,
    unsigned long long stride0)
{
    unsigned long long ga = (unsigned long long)(uintptr_t)gptr;
    v4u g0;
    g0[0] = 1u;                                            // count=1, user mode
    g0[1] = lds_byte_off;                                  // lds_addr
    g0[2] = (unsigned)ga;                                  // global_addr[31:0]
    g0[3] = (unsigned)((ga >> 32) & 0x1FFFFFFu)            // global_addr[56:32]
          | (2u << 30);                                    // type=2 ("image")
    v8u g1;
    g1[0] = 0x00010000u;                                   // wg_mask=0, data_size=1 (2B)
    g1[1] = (tensor_d0 & 0xFFFFu) << 16;                   // tensor_dim0[15:0]
    g1[2] = (tensor_d0 >> 16) | ((tensor_d1 & 0xFFFFu) << 16);
    g1[3] = (tensor_d1 >> 16) | (tile_d0 << 16);           // tile_dim0
    g1[4] = tile_d1 & 0xFFFFu;                             // tile_dim1, tile_dim2=0
    g1[5] = (unsigned)stride0;                             // tensor_dim0_stride[31:0]
    g1[6] = (unsigned)(stride0 >> 32) & 0xFFFFu;           // stride0[47:32], stride1=0
    g1[7] = 0u;
    asm volatile("tensor_load_to_lds %0, %1" :: "s"(g0), "s"(g1) : "memory");
}

// ---------------- kernel 0: fp32 -> f16 weight convert ----------------
__global__ __launch_bounds__(256) void enc_cvt_f16(const float* __restrict__ src,
                                                   _Float16* __restrict__ dst, int n) {
    int i = blockIdx.x * blockDim.x + threadIdx.x;
    if (i < n) dst[i] = (_Float16)src[i];
}

// ---------------- kernel 1: fused LayerNorm + QKV projection ----------------
// grid = NROWS/16 blocks, 256 threads (8 waves). Each block: 16 rows.
// Q,K stored row-major f16 [NROWS][ENC]; V stored transposed [ENC][NROWS].
// Q pre-scaled by 1/sqrt(ENC) = 1/16 (exact in f16).
__global__ __launch_bounds__(256) void enc_ln_qkv(
    const float* __restrict__ x, const float* __restrict__ gamma, const float* __restrict__ beta,
    const float* __restrict__ bQ, const float* __restrict__ bK, const float* __restrict__ bV,
    const _Float16* __restrict__ wq16, const _Float16* __restrict__ wk16,
    const _Float16* __restrict__ wv16,
    _Float16* __restrict__ q16, _Float16* __restrict__ k16, _Float16* __restrict__ vt16)
{
    __shared__ _Float16 xn[16][272];     // padded row stride (544B, 16B multiple)
    __shared__ float rsum[16][16];
    __shared__ float rsq[16][16];

    const int t = threadIdx.x;
    const int rowbase = blockIdx.x * 16;
    const int r  = t >> 4;
    const int cs = (t & 15) << 4;

    // ---- LayerNorm: 16 threads cooperate per row, 16 values each ----
    const float* xp = x + (size_t)(rowbase + r) * EMB + cs;
    float vals[16];
#pragma unroll
    for (int j = 0; j < 4; j++) {
        float4 f = ((const float4*)xp)[j];
        vals[4*j+0] = f.x; vals[4*j+1] = f.y; vals[4*j+2] = f.z; vals[4*j+3] = f.w;
    }
    float s = 0.f, sq = 0.f;
#pragma unroll
    for (int j = 0; j < 16; j++) { s += vals[j]; sq += vals[j] * vals[j]; }
    rsum[r][t & 15] = s;
    rsq [r][t & 15] = sq;
    __syncthreads();

    float fs = 0.f, fq = 0.f;
#pragma unroll
    for (int j = 0; j < 16; j++) { fs += rsum[r][j]; fq += rsq[r][j]; }
    const float mu   = fs * (1.0f / EMB);
    const float var  = fq * (1.0f / EMB) - mu * mu;
    const float rstd = rsqrtf(var + 1e-5f);

    union { _Float16 h[16]; uint4 q[2]; } nb;
#pragma unroll
    for (int j = 0; j < 16; j++)
        nb.h[j] = (_Float16)((vals[j] - mu) * rstd * gamma[cs + j] + beta[cs + j]);
    *(uint4*)&xn[r][cs]     = nb.q[0];
    *(uint4*)&xn[r][cs + 8] = nb.q[1];
    __syncthreads();

    // ---- QKV via WMMA: 48 output 16x16 tiles, 6 per wave ----
    const int wave = t >> 5, lane = t & 31;
    const int col  = lane & 15;
    const int half = lane >> 4;

    F16Frag afr[8];
#pragma unroll
    for (int kc = 0; kc < 8; kc++) {
        const _Float16* p = &xn[col][kc * 32];
        afr[kc].q[0] = *(const uint4*)(p + 16 * half);
        afr[kc].q[1] = *(const uint4*)(p + 16 * half + 8);
    }

#pragma unroll
    for (int j = 0; j < 6; j++) {
        const int T  = wave * 6 + j;           // 0..47
        const int m  = T >> 4;                 // 0=Q 1=K 2=V
        const int n0 = (T & 15) << 4;
        const _Float16* W    = (m == 0) ? wq16 : (m == 1) ? wk16 : wv16;
        const float*    bias = (m == 0) ? bQ   : (m == 1) ? bK   : bV;

        const float bv = bias[n0 + col];
        v8f acc;
#pragma unroll
        for (int i = 0; i < 8; i++) acc[i] = bv;

#pragma unroll
        for (int kc = 0; kc < 8; kc++) {
            F16Frag bf;  // B[k][n] = W[n][k] : contiguous along k
            const _Float16* p = W + (size_t)(n0 + col) * EMB + kc * 32 + 16 * half;
            bf.q[0] = *(const uint4*)(p);
            bf.q[1] = *(const uint4*)(p + 8);
            acc = wmma_f16(afr[kc], bf, acc);
        }

        if (m < 2) {
            _Float16* dst = (m == 0) ? q16 : k16;
            const float sc = (m == 0) ? 0.0625f : 1.0f;   // 1/sqrt(256) folded into Q
#pragma unroll
            for (int i = 0; i < 8; i++)
                dst[(size_t)(rowbase + i + 8 * half) * ENC + n0 + col] = (_Float16)(acc[i] * sc);
        } else {
#pragma unroll
            for (int i = 0; i < 8; i++)
                vt16[(size_t)(n0 + col) * NROWS + rowbase + i + 8 * half] = (_Float16)acc[i];
        }
    }
}

// ---------------- kernel 2: flash attention + output projection ----------------
// 128 threads / block, 4 waves; each wave owns one 16-query tile; the block's
// 32-key K/V tiles are TDM-loaded into LDS and shared by all 4 waves.
__global__ __launch_bounds__(128) void enc_attn_out(
    const _Float16* __restrict__ q16, const _Float16* __restrict__ k16,
    const _Float16* __restrict__ vt16, const _Float16* __restrict__ w016,
    const float* __restrict__ b0, float* __restrict__ out)
{
    // smem arena: during key loop: kls [32][256] f16 (16KB) + vls [256][32] f16 (16KB)
    // after the loop the same arena holds hbuf[4][16][272] f16 (34816B).
    __shared__ __align__(16) char smem[36864];
    __shared__ _Float16 pbuf[4][16][40];

    _Float16 (*kls)[256] = (_Float16 (*)[256])(smem);
    _Float16 (*vls)[32]  = (_Float16 (*)[32])(smem + 16384);

    const int t = threadIdx.x, wave = t >> 5, lane = t & 31;
    const int tile = blockIdx.x * 4 + wave;           // 0..1023
    const int b    = tile >> 8;                       // 256 tiles per batch
    const int q0   = (tile & 255) << 4;
    const int rowg = b * SEQ + q0;
    const int col  = lane & 15;
    const int half = lane >> 4;

    // Q resident in registers (A-fragments, 16x32 per k-chunk)
    F16Frag qf[8];
#pragma unroll
    for (int kc = 0; kc < 8; kc++) {
        const _Float16* p = q16 + (size_t)(rowg + col) * ENC + kc * 32;
        qf[kc].q[0] = *(const uint4*)(p + 16 * half);
        qf[kc].q[1] = *(const uint4*)(p + 16 * half + 8);
    }

    v8f o[16];
#pragma unroll
    for (int n = 0; n < 16; n++)
#pragma unroll
        for (int i = 0; i < 8; i++) o[n][i] = 0.f;

    float mrow[8], lrow[8];
#pragma unroll
    for (int i = 0; i < 8; i++) { mrow[i] = -1e30f; lrow[i] = 0.f; }

    // block-shared key range: wave0's tile start (multiple of 64, hence of 32)
    const int q0base = ((blockIdx.x * 4) & 255) << 4;

    for (int kb = q0base; kb < SEQ; kb += 32) {
        __syncthreads();                       // previous tile consumed, LDS free
        if (wave == 0) {
            // K tile: 32 rows x 256 enc, row-major, stride ENC
            tdm_load_2d_f16(lds_off_of(&kls[0][0]),
                            k16 + (size_t)(b * SEQ + kb) * ENC,
                            ENC, (unsigned)(NROWS - (b * SEQ + kb)),
                            ENC, 32, (unsigned long long)ENC);
            // V^T tile: 256 rows x 32 keys, from vt16 (stride NROWS)
            tdm_load_2d_f16(lds_off_of(&vls[0][0]),
                            vt16 + (size_t)(b * SEQ + kb),
                            32, ENC, 32, ENC, (unsigned long long)NROWS);
            __builtin_amdgcn_s_wait_tensorcnt(0);
        }
        __syncthreads();                       // tiles visible to all waves

        if (kb + 32 > q0) {                    // skip fully-masked blocks for this wave
            // ---- scores S = Q K^T (Q pre-scaled), two 16x16 tiles ----
            v8f s0, s1;
#pragma unroll
            for (int i = 0; i < 8; i++) { s0[i] = 0.f; s1[i] = 0.f; }

#pragma unroll
            for (int kc = 0; kc < 8; kc++) {
                F16Frag bk0, bk1;
                const _Float16* kp0 = &kls[col][kc * 32 + 16 * half];
                bk0.q[0] = *(const uint4*)(kp0);
                bk0.q[1] = *(const uint4*)(kp0 + 8);
                const _Float16* kp1 = &kls[16 + col][kc * 32 + 16 * half];
                bk1.q[0] = *(const uint4*)(kp1);
                bk1.q[1] = *(const uint4*)(kp1 + 8);
                s0 = wmma_f16(qf[kc], bk0, s0);
                s1 = wmma_f16(qf[kc], bk1, s1);
            }

            // ---- reverse-causal mask (attend to k >= q) ----
            if (kb < q0 + 16) {
#pragma unroll
                for (int i = 0; i < 8; i++) {
                    const int qrow = q0 + i + 8 * half;
                    if (kb + col      < qrow) s0[i] = -1e30f;
                    if (kb + 16 + col < qrow) s1[i] = -1e30f;
                }
            }

            // ---- online softmax (row M = i + 8*half across 16-lane group) ----
#pragma unroll
            for (int i = 0; i < 8; i++) {
                float m2 = fmaxf(s0[i], s1[i]);
#pragma unroll
                for (int d = 1; d < 16; d <<= 1) m2 = fmaxf(m2, __shfl_xor(m2, d));
                const float mn   = fmaxf(mrow[i], m2);
                const float corr = __expf(mrow[i] - mn);
                mrow[i] = mn;
                const float p0 = __expf(s0[i] - mn);
                const float p1 = __expf(s1[i] - mn);
                float rs = p0 + p1;
#pragma unroll
                for (int d = 1; d < 16; d <<= 1) rs += __shfl_xor(rs, d);
                lrow[i] = lrow[i] * corr + rs;
#pragma unroll
                for (int n = 0; n < 16; n++) o[n][i] *= corr;
                pbuf[wave][i + 8 * half][col]      = (_Float16)p0;
                pbuf[wave][i + 8 * half][16 + col] = (_Float16)p1;
            }
            asm volatile("s_wait_dscnt 0" ::: "memory");

            // ---- O += P V : P as 16x32 A-fragment, V^T tile as B-fragments ----
            F16Frag pa;
            const _Float16* pp = &pbuf[wave][col][0];
            pa.q[0] = *(const uint4*)(pp + 16 * half);
            pa.q[1] = *(const uint4*)(pp + 16 * half + 8);

#pragma unroll
            for (int n = 0; n < 16; n++) {
                F16Frag bv;  // B[k][n] = V[kb+k][n*16+col] = vls[n*16+col][k]
                const _Float16* vp = &vls[n * 16 + col][16 * half];
                bv.q[0] = *(const uint4*)(vp);
                bv.q[1] = *(const uint4*)(vp + 8);
                o[n] = wmma_f16(pa, bv, o[n]);
            }
        }
    }

    __syncthreads();   // all waves done with kls/vls before aliasing as hbuf

    // ---- normalize O, stage f16, out = h @ W0^T + b0 ----
    _Float16 (*hb)[272] = (_Float16 (*)[272])(smem + wave * 16 * 272 * 2);

#pragma unroll
    for (int i = 0; i < 8; i++) {
        const float rl = 1.0f / lrow[i];
#pragma unroll
        for (int n = 0; n < 16; n++)
            hb[i + 8 * half][n * 16 + col] = (_Float16)(o[n][i] * rl);
    }
    asm volatile("s_wait_dscnt 0" ::: "memory");

    F16Frag ha[8];
#pragma unroll
    for (int kc = 0; kc < 8; kc++) {
        const _Float16* p = &hb[col][kc * 32];
        ha[kc].q[0] = *(const uint4*)(p + 16 * half);
        ha[kc].q[1] = *(const uint4*)(p + 16 * half + 8);
    }

#pragma unroll
    for (int n = 0; n < 16; n++) {
        const float bb = b0[n * 16 + col];
        v8f acc;
#pragma unroll
        for (int i = 0; i < 8; i++) acc[i] = bb;
#pragma unroll
        for (int kc = 0; kc < 8; kc++) {
            F16Frag wb;  // B[k][d] = w0[d][k], contiguous along k
            const _Float16* p = w016 + (size_t)(n * 16 + col) * ENC + kc * 32 + 16 * half;
            wb.q[0] = *(const uint4*)(p);
            wb.q[1] = *(const uint4*)(p + 8);
            acc = wmma_f16(ha[kc], wb, acc);
        }
#pragma unroll
        for (int i = 0; i < 8; i++)
            out[(size_t)(b * SEQ + q0 + i + 8 * half) * EMB + n * 16 + col] = acc[i];
    }
}

// ---------------- launcher ----------------
extern "C" void kernel_launch(void* const* d_in, const int* in_sizes, int n_in,
                              void* d_out, int out_size, void* d_ws, size_t ws_size,
                              hipStream_t stream) {
    const float* x        = (const float*)d_in[0];
    const float* ln_gamma = (const float*)d_in[1];
    const float* ln_beta  = (const float*)d_in[2];
    const float* wQ       = (const float*)d_in[3];
    const float* bQ       = (const float*)d_in[4];
    const float* wK       = (const float*)d_in[5];
    const float* bK       = (const float*)d_in[6];
    const float* wV       = (const float*)d_in[7];
    const float* bV       = (const float*)d_in[8];
    const float* w0       = (const float*)d_in[9];
    const float* b0       = (const float*)d_in[10];
    float* out = (float*)d_out;

    // workspace layout (f16 elements)
    const size_t NQ = (size_t)NROWS * ENC;          // 4M elems each
    _Float16* w    = (_Float16*)d_ws;
    _Float16* q16  = w;
    _Float16* k16  = w + NQ;
    _Float16* vt16 = w + 2 * NQ;                    // transposed [ENC][NROWS]
    _Float16* wq16 = w + 3 * NQ;
    _Float16* wk16 = wq16 + WELEM;
    _Float16* wv16 = wk16 + WELEM;
    _Float16* w016 = wv16 + WELEM;

    enc_cvt_f16<<<WELEM / 256, 256, 0, stream>>>(wQ, wq16, WELEM);
    enc_cvt_f16<<<WELEM / 256, 256, 0, stream>>>(wK, wk16, WELEM);
    enc_cvt_f16<<<WELEM / 256, 256, 0, stream>>>(wV, wv16, WELEM);
    enc_cvt_f16<<<WELEM / 256, 256, 0, stream>>>(w0, w016, WELEM);

    enc_ln_qkv<<<NROWS / 16, 256, 0, stream>>>(x, ln_gamma, ln_beta, bQ, bK, bV,
                                               wq16, wk16, wv16, q16, k16, vt16);

    enc_attn_out<<<(BATCH * SEQ / 16) / 4, 128, 0, stream>>>(q16, k16, vt16, w016, b0, out);
}
